// VmFusionBlocker_78348793414350
// MI455X (gfx1250) — compile-verified, matrix operand order checked by reading, not run on hardware
//
#include <hip/hip_runtime.h>

// ---------------------------------------------------------------------------
// Types for WMMA fragments (gfx1250, wave32)
// ---------------------------------------------------------------------------
typedef __attribute__((ext_vector_type(16))) _Float16 v16h;
typedef __attribute__((ext_vector_type(8)))  float    v8f;

// Problem constants
#define MTOK   2000        // 4 * 500 tokens
#define DMODEL 1024
#define DINNER 2048
#define DSTATE 16
#define DTRANK 64
#define SEQL   500
#define NBATCH 4

// GEMM tiling: block = 256 threads = 8 waves; block tile 64x128;
// each wave computes a 32x32 output tile = 4 WMMA accumulators.
#define BM 64
#define BN 128
#define BK 32

__device__ __forceinline__ float silu_f(float x) {
    return x / (1.0f + __expf(-x));
}

// ---------------------------------------------------------------------------
// CDNA5 async global->LDS copy (ASYNCcnt-tracked DMA, bypasses VGPRs).
// LDS byte offset = low 32 bits of the flat pointer (LDS aperture layout).
// ---------------------------------------------------------------------------
__device__ __forceinline__ void async_ld_b128(void* lds_dst, const float* gsrc) {
    unsigned off = (unsigned)(size_t)lds_dst;
    asm volatile("global_load_async_to_lds_b128 %0, %1, off"
                 :: "v"(off), "v"(gsrc)
                 : "memory");
}
__device__ __forceinline__ void wait_async_all() {
    asm volatile("s_wait_asynccnt 0x0" ::: "memory");
}

// Build a 16-half A/B fragment from an fp32 LDS row.
// Per documented 16-bit layout: lane&15 = row/col, K chunks at
// kbase=(lane>>4)*8 and kbase+16.
__device__ __forceinline__ v16h frag_from_lds(const float* rowp, int kbase) {
    v8f f0 = *(const v8f*)(rowp + kbase);
    v8f f1 = *(const v8f*)(rowp + kbase + 16);
    v16h r;
    #pragma unroll
    for (int i = 0; i < 8; ++i) {
        r[i]     = (_Float16)f0[i];
        r[i + 8] = (_Float16)f1[i];
    }
    return r;
}

// ---------------------------------------------------------------------------
// WMMA GEMM:  C[M x N] = epilogue(A[M x K] @ W[N x K]^T + bias)
// A row stride = lda, W row stride = K, C row stride = ldc.
// EPI: 0 = none, 1 = softplus(x + bias), 2 = tanh(x + bias)
// OOB staging rows are CLAMPED (never branch): clamped garbage only feeds
// output rows/cols that the guarded epilogue never stores.
// K must be a multiple of 32 (true for all call sites here).
// ---------------------------------------------------------------------------
template<int EPI>
__global__ __launch_bounds__(256)
void gemm_wmma_f16(const float* __restrict__ A, int lda,
                   const float* __restrict__ W,
                   const float* __restrict__ bias,
                   float* __restrict__ C, int ldc,
                   int M, int N, int K)
{
    __shared__ __align__(32) float As[BM][BK];   // 8 KB (fp32: async copy can't convert)
    __shared__ __align__(32) float Ws[BN][BK];   // 16 KB

    const int tid  = threadIdx.x;
    const int lane = tid & 31;
    const int wave = tid >> 5;        // 0..7
    const int wm   = wave & 1;        // m 32-row half of block tile
    const int wn   = wave >> 1;       // n 32-col quarter of block tile
    const int bm0  = blockIdx.x * BM;
    const int bn0  = blockIdx.y * BN;

    const int l15   = lane & 15;
    const int kbase = (lane >> 4) * 8;

    v8f acc00 = {}, acc01 = {}, acc10 = {}, acc11 = {};

    for (int k0 = 0; k0 < K; k0 += BK) {
        // ---- async stage A tile (64x32 fp32): 2 x b128 per thread ---------
        #pragma unroll
        for (int i = 0; i < 2; ++i) {
            const int e   = tid + i * 256;     // 0..511
            const int row = e >> 3;            // 0..63
            const int kc  = (e & 7) * 4;       // 0..28
            int gr = bm0 + row; if (gr > M - 1) gr = M - 1;   // clamp
            async_ld_b128(&As[row][kc], A + (size_t)gr * lda + k0 + kc);
        }
        // ---- async stage W tile (128x32 fp32): 4 x b128 per thread --------
        #pragma unroll
        for (int i = 0; i < 4; ++i) {
            const int e   = tid + i * 256;     // 0..1023
            const int row = e >> 3;            // 0..127
            const int kc  = (e & 7) * 4;
            int gr = bn0 + row; if (gr > N - 1) gr = N - 1;   // clamp
            async_ld_b128(&Ws[row][kc], W + (size_t)gr * K + k0 + kc);
        }
        wait_async_all();       // s_wait_asynccnt 0 (per-wave)
        __syncthreads();        // publish LDS tile to all waves

        // ---- fragments (f32 LDS -> f16 regs) + 4 WMMAs with A/B reuse -----
        v16h a0 = frag_from_lds(&As[wm * 32 +      l15][0], kbase);
        v16h a1 = frag_from_lds(&As[wm * 32 + 16 + l15][0], kbase);
        v16h b0 = frag_from_lds(&Ws[wn * 32 +      l15][0], kbase);
        v16h b1 = frag_from_lds(&Ws[wn * 32 + 16 + l15][0], kbase);

        acc00 = __builtin_amdgcn_wmma_f32_16x16x32_f16(false, a0, false, b0, (short)0, acc00, false, false);
        acc01 = __builtin_amdgcn_wmma_f32_16x16x32_f16(false, a0, false, b1, (short)0, acc01, false, false);
        acc10 = __builtin_amdgcn_wmma_f32_16x16x32_f16(false, a1, false, b0, (short)0, acc10, false, false);
        acc11 = __builtin_amdgcn_wmma_f32_16x16x32_f16(false, a1, false, b1, (short)0, acc11, false, false);

        __syncthreads();        // tile consumed; safe to overwrite next iter
    }

    // ---- epilogue + guarded store -----------------------------------------
    #pragma unroll
    for (int im = 0; im < 2; ++im) {
        #pragma unroll
        for (int jn = 0; jn < 2; ++jn) {
            const v8f acc = (im == 0) ? (jn == 0 ? acc00 : acc01)
                                      : (jn == 0 ? acc10 : acc11);
            const int col = bn0 + wn * 32 + jn * 16 + l15;
            const int rb  = bm0 + wm * 32 + im * 16 + (lane >> 4) * 8;
            float bv = 0.0f;
            if (EPI != 0 && col < N) bv = bias[col];
            #pragma unroll
            for (int j = 0; j < 8; ++j) {
                const int r = rb + j;
                if (r < M && col < N) {
                    float v = acc[j];
                    if (EPI == 1) {                 // softplus(x + b)
                        v += bv;
                        v = (v > 20.0f) ? v : log1pf(__expf(v));
                    } else if (EPI == 2) {          // tanh(x + b)
                        v = tanhf(v + bv);
                    }
                    C[(size_t)r * ldc + col] = v;
                }
            }
        }
    }
}

// ---------------------------------------------------------------------------
// Causal depthwise conv (width 4) + SiLU. One thread per (batch, channel).
// ---------------------------------------------------------------------------
__global__ __launch_bounds__(256)
void conv_silu_kernel(const float* __restrict__ xz,
                      const float* __restrict__ conv_w,
                      const float* __restrict__ conv_b,
                      float* __restrict__ xc)
{
    const int t = blockIdx.x * blockDim.x + threadIdx.x;
    if (t >= NBATCH * DINNER) return;
    const int b = t >> 11;
    const int d = t & (DINNER - 1);

    const float w0 = conv_w[d * 4 + 0];
    const float w1 = conv_w[d * 4 + 1];
    const float w2 = conv_w[d * 4 + 2];
    const float w3 = conv_w[d * 4 + 3];
    const float cb = conv_b[d];

    const float* __restrict__ in  = xz + (size_t)b * SEQL * (2 * DINNER) + d;
    float*       __restrict__ out = xc + (size_t)b * SEQL * DINNER + d;

    float x0 = 0.f, x1 = 0.f, x2 = 0.f;
    for (int l = 0; l < SEQL; ++l) {
        const float x3 = in[(size_t)l * (2 * DINNER)];
        const float s  = w0 * x0 + w1 * x1 + w2 * x2 + w3 * x3 + cb;
        out[(size_t)l * DINNER] = silu_f(s);
        x0 = x1; x1 = x2; x2 = x3;
    }
}

// ---------------------------------------------------------------------------
// Selective scan. One thread per (batch, channel): 500 sequential steps,
// 16 SSM states in registers.
// ---------------------------------------------------------------------------
__global__ __launch_bounds__(256)
void ssm_scan_kernel(const float* __restrict__ delta,
                     const float* __restrict__ x_dbl,   // (B,L,96): dt|B|C
                     const float* __restrict__ xc,
                     const float* __restrict__ xz,      // z = cols [DINNER, 2*DINNER)
                     const float* __restrict__ A_log,
                     const float* __restrict__ Dp,
                     float* __restrict__ y)
{
    const int t = blockIdx.x * blockDim.x + threadIdx.x;
    if (t >= NBATCH * DINNER) return;
    const int b = t >> 11;
    const int d = t & (DINNER - 1);

    float a[DSTATE], h[DSTATE];
    #pragma unroll
    for (int n = 0; n < DSTATE; ++n) {
        a[n] = -__expf(A_log[d * DSTATE + n]);
        h[n] = 0.f;
    }
    const float Dd = Dp[d];

    const float* __restrict__ drow = delta + (size_t)b * SEQL * DINNER + d;
    const float* __restrict__ urow = xc    + (size_t)b * SEQL * DINNER + d;
    const float* __restrict__ zrow = xz    + (size_t)b * SEQL * (2 * DINNER) + DINNER + d;
    const float* __restrict__ xdb  = x_dbl + (size_t)b * SEQL * 96;
    float*       __restrict__ yrow = y     + (size_t)b * SEQL * DINNER + d;

    for (int l = 0; l < SEQL; ++l) {
        const float dl = drow[(size_t)l * DINNER];
        const float u  = urow[(size_t)l * DINNER];
        const float du = dl * u;
        const float* __restrict__ Bl = xdb + (size_t)l * 96 + DTRANK;
        const float* __restrict__ Cl = xdb + (size_t)l * 96 + DTRANK + DSTATE;
        float acc = 0.f;
        #pragma unroll
        for (int n = 0; n < DSTATE; ++n) {
            const float dA = __expf(dl * a[n]);
            h[n] = dA * h[n] + du * Bl[n];
            acc += h[n] * Cl[n];
        }
        const float z = zrow[(size_t)l * (2 * DINNER)];
        yrow[(size_t)l * DINNER] = (acc + u * Dd) * silu_f(z);
    }
}

// ---------------------------------------------------------------------------
// Host-side orchestration
// ---------------------------------------------------------------------------
extern "C" void kernel_launch(void* const* d_in, const int* in_sizes, int n_in,
                              void* d_out, int out_size, void* d_ws, size_t ws_size,
                              hipStream_t stream) {
    const float* x        = (const float*)d_in[0];   // (2000,1024)
    const float* in_proj  = (const float*)d_in[1];   // (4096,1024)
    const float* conv_w   = (const float*)d_in[2];   // (2048,4)
    const float* conv_b   = (const float*)d_in[3];   // (2048)
    const float* x_proj   = (const float*)d_in[4];   // (96,2048)
    const float* dt_proj  = (const float*)d_in[5];   // (2048,64)
    const float* dt_b     = (const float*)d_in[6];   // (2048)
    const float* A_log    = (const float*)d_in[7];   // (2048,16)
    const float* Dp       = (const float*)d_in[8];   // (2048)
    const float* out_proj = (const float*)d_in[9];   // (1024,2048)
    const float* fc1_w    = (const float*)d_in[10];  // (1024,10240)
    const float* fc1_b    = (const float*)d_in[11];  // (1024)
    float* out = (float*)d_out;

    // Workspace layout (floats)
    const size_t n_xz    = (size_t)MTOK * 2 * DINNER;
    const size_t n_xc    = (size_t)MTOK * DINNER;
    const size_t n_xdbl  = (size_t)MTOK * 96;
    const size_t n_delta = (size_t)MTOK * DINNER;
    const size_t n_y     = (size_t)MTOK * DINNER;
    if (ws_size < (n_xz + n_xc + n_xdbl + n_delta + n_y) * sizeof(float)) return;

    float* xz    = (float*)d_ws;
    float* xc    = xz    + n_xz;
    float* xdbl  = xc    + n_xc;
    float* delta = xdbl  + n_xdbl;
    float* yb    = delta + n_delta;
    float* xm    = xz;   // reuse: z no longer needed after the scan

    const dim3 blk(256);
    const int  mt = (MTOK + BM - 1) / BM;   // 32

    // 1) xz = X @ in_proj^T            (2000 x 4096, K=1024)
    gemm_wmma_f16<0><<<dim3(mt, (2 * DINNER) / BN), blk, 0, stream>>>(
        x, DMODEL, in_proj, nullptr, xz, 2 * DINNER, MTOK, 2 * DINNER, DMODEL);

    // 2) causal conv(4) + SiLU -> xc
    conv_silu_kernel<<<(NBATCH * DINNER) / 256, blk, 0, stream>>>(xz, conv_w, conv_b, xc);

    // 3) x_dbl = xc @ x_proj^T         (2000 x 96, K=2048)
    gemm_wmma_f16<0><<<dim3(mt, 1), blk, 0, stream>>>(
        xc, DINNER, x_proj, nullptr, xdbl, 96, MTOK, 96, DINNER);

    // 4) delta = softplus(dt @ dt_proj^T + b)   (2000 x 2048, K=64, lda=96)
    gemm_wmma_f16<1><<<dim3(mt, DINNER / BN), blk, 0, stream>>>(
        xdbl, 96, dt_proj, dt_b, delta, DINNER, MTOK, DINNER, DTRANK);

    // 5) selective scan -> yb
    ssm_scan_kernel<<<(NBATCH * DINNER) / 256, blk, 0, stream>>>(
        delta, xdbl, xc, xz, A_log, Dp, yb);

    // 6) xm = yb @ out_proj^T          (2000 x 1024, K=2048)
    gemm_wmma_f16<0><<<dim3(mt, DMODEL / BN), blk, 0, stream>>>(
        yb, DINNER, out_proj, nullptr, xm, DMODEL, MTOK, DMODEL, DINNER);

    // 7) out = tanh(xm(200x10240) @ fc1^T + b)  (200 x 1024, K=10240)
    gemm_wmma_f16<2><<<dim3((200 + BM - 1) / BM, DMODEL / BN), blk, 0, stream>>>(
        xm, 10240, fc1_w, fc1_b, out, DMODEL, 200, DMODEL, 10240);
}
